// ShapeDecoder_48464410968236
// MI455X (gfx1250) — matrix operand order repaired and checked by reading, not compiled, not run on hardware
//
#include <hip/hip_runtime.h>
#include <hip/hip_bf16.h>

typedef __attribute__((ext_vector_type(16))) __bf16 v16bf;
typedef __attribute__((ext_vector_type(8)))  float  v8f;

#define TBF       128            // frames per block (4 waves of 32)
#define NPTS      1000
#define NT_TOTAL  63             // ceil(1000/16)
#define W9_ROWS_PAD 1008         // 63*16, padded rows zero-filled
#define NWAVES    (TBF / 32)
#define W9_HALF_BYTES (W9_ROWS_PAD * 32 * 2)   // 64512 B per hi/lo array (16B aligned)

struct MlpW { const float* w[8]; const float* b[8]; };

// round-to-nearest-even f32 -> bf16 bits (inputs finite)
__device__ __forceinline__ unsigned bf_round(float f) {
    unsigned u = __float_as_uint(f);
    u += 0x7FFFu + ((u >> 16) & 1u);
    return u >> 16;
}
// split f32 into hi/lo bf16 bit patterns: f ~= bf(hi) + bf(lo)
__device__ __forceinline__ void bf_split(float f, unsigned& hi, unsigned& lo) {
    hi = bf_round(f);
    float fhi = __uint_as_float(hi << 16);
    lo = bf_round(f - fhi);
}

// Dense layer + ELU. Inputs/outputs live in registers (compile-time indexed);
// the dynamically-indexed output side round-trips through a padded per-thread
// LDS slot (stride 33 floats -> all 32 lanes hit distinct banks).
template<int FIN, int FOUT>
__device__ __forceinline__ void dense_elu(const float* __restrict__ sw,
                                          const float* __restrict__ sb,
                                          const float* in, float* outp,
                                          volatile float* myact) {
#pragma unroll 1
    for (int o = 0; o < FOUT; ++o) {
        float acc = sb[o];
        const float* wr = sw + o * FIN;   // uniform address -> LDS broadcast
#pragma unroll
        for (int i = 0; i < FIN; ++i) acc = fmaf(wr[i], in[i], acc);
        acc = (acc > 0.f) ? acc : (__expf(acc) - 1.f);   // ELU (alpha=1)
        myact[o] = acc;
    }
#pragma unroll
    for (int o = 0; o < FOUT; ++o) outp[o] = myact[o];
}

__global__ __launch_bounds__(TBF) void prep_w9_kernel(const float* __restrict__ W9,
                                                      unsigned short* __restrict__ w9hi,
                                                      unsigned short* __restrict__ w9lo) {
    int i = blockIdx.x * TBF + threadIdx.x;
    if (i < W9_ROWS_PAD * 32) {
        int n = i >> 5;
        float v = (n < NPTS) ? W9[i] : 0.f;
        unsigned hi, lo;
        bf_split(v, hi, lo);
        w9hi[i] = (unsigned short)hi;
        w9lo[i] = (unsigned short)lo;
    }
}

__global__ __launch_bounds__(TBF) void shape_decoder_kernel(
    const float* __restrict__ x, const float* __restrict__ init_shape,
    MlpW P, const uint4* __restrict__ w9hq, const uint4* __restrict__ w9lq,
    float* __restrict__ out, int nframes)
{
    __shared__ float sW[2972];            // 2834 weights + 138 biases
    __shared__ float sAct[TBF * 33];      // per-thread activation slots (padded)
    __shared__ uint4 sHhi[TBF * 4];       // hidden bf16 hi: 4 x 16B chunks per frame
    __shared__ uint4 sHlo[TBF * 4];       // hidden bf16 lo

    const int tid = threadIdx.x;

    // ---- stage all fc1..fc8 weights/biases into LDS ----
    {
        constexpr int wsz[8] = {2, 16, 64, 64, 128, 512, 1024, 1024};
        constexpr int bsz[8] = {2, 8, 8, 8, 16, 32, 32, 32};
        int off = 0;
#pragma unroll
        for (int l = 0; l < 8; ++l) {
            for (int j = tid; j < wsz[l]; j += TBF) sW[off + j] = P.w[l][j];
            off += wsz[l];
        }
#pragma unroll
        for (int l = 0; l < 8; ++l) {
            for (int j = tid; j < bsz[l]; j += TBF) sW[off + j] = P.b[l][j];
            off += bsz[l];
        }
    }
    __syncthreads();

    // ---- per-thread MLP: one frame per thread ----
    const int base  = blockIdx.x * TBF;
    const int frame = base + tid;
    float xin = (frame < nframes) ? x[frame] : 0.f;
    volatile float* myact = &sAct[tid * 33];

    float ha[32], hb[32];
    ha[0] = xin;
    // weight offsets: 0,2,18,82,146,274,786,1810 ; bias offsets from 2834
    dense_elu<1,  2 >(sW + 0,    sW + 2834, ha, hb, myact);
    dense_elu<2,  8 >(sW + 2,    sW + 2836, hb, ha, myact);
    dense_elu<8,  8 >(sW + 18,   sW + 2844, ha, hb, myact);
    dense_elu<8,  8 >(sW + 82,   sW + 2852, hb, ha, myact);
    dense_elu<8,  16>(sW + 146,  sW + 2860, ha, hb, myact);
    dense_elu<16, 32>(sW + 274,  sW + 2876, hb, ha, myact);
    dense_elu<32, 32>(sW + 786,  sW + 2908, ha, hb, myact);
    dense_elu<32, 32>(sW + 1810, sW + 2940, hb, ha, myact);
    // ha[0..31] = final hidden vector

    // split to hi/lo bf16, store rows: chunks K0-7 | K8-15 | K16-23 | K24-31
#pragma unroll
    for (int c = 0; c < 4; ++c) {
        unsigned h[8], l[8];
#pragma unroll
        for (int e = 0; e < 8; ++e) bf_split(ha[c * 8 + e], h[e], l[e]);
        uint4 qh, ql;
        qh.x = h[0] | (h[1] << 16);  ql.x = l[0] | (l[1] << 16);
        qh.y = h[2] | (h[3] << 16);  ql.y = l[2] | (l[3] << 16);
        qh.z = h[4] | (h[5] << 16);  ql.z = l[4] | (l[5] << 16);
        qh.w = h[6] | (h[7] << 16);  ql.w = l[6] | (l[7] << 16);
        sHhi[tid * 4 + c] = qh;
        sHlo[tid * 4 + c] = ql;
    }
    __syncthreads();

    // ---- GEMM: (Hhi+Hlo) @ (W9hi+W9lo)^T + init_shape, 3-term bf16 split ----
    const int lane   = tid & 31;
    const int wave   = tid >> 5;
    const int lid    = lane & 15;
    const int half16 = lane >> 4;      // 0: lanes 0-15, 1: lanes 16-31

    int vframes = nframes - base; if (vframes > TBF) vframes = TBF;
    const int mtc = vframes >> 4;      // nframes % 16 == 0 -> all tiles full

    for (int nt = wave; nt < NT_TOTAL; nt += NWAVES) {
        const int col = nt * 16 + lid;
        const bool cok = (col < NPTS);
        const float sv = cok ? init_shape[col] : 0.f;

        // B operand: lanes 0-15 hold col, K0-15; lanes 16-31 hold K16-31
        const int bbase = col * 4 + half16 * 2;
        union { v16bf v; uint4 q[2]; } Bh, Bl;
        Bh.q[0] = w9hq[bbase + 0];
        Bh.q[1] = w9hq[bbase + 1];
        Bl.q[0] = w9lq[bbase + 0];
        Bl.q[1] = w9lq[bbase + 1];

        for (int mt = 0; mt < mtc; ++mt) {
            const int row = mt * 16 + lid;
            // A operand: lanes 0-15: K0-15 ; lanes 16-31: K16-31
            const int abase = row * 4 + half16 * 2;
            union { v16bf v; uint4 q[2]; } Ah, Al;
            Ah.q[0] = sHhi[abase];
            Ah.q[1] = sHhi[abase + 1];
            Al.q[0] = sHlo[abase];
            Al.q[1] = sHlo[abase + 1];

            v8f c;
#pragma unroll
            for (int r = 0; r < 8; ++r) c[r] = sv;   // bias seeded via C operand

            // small terms first, fp32 accumulation throughout
            c = __builtin_amdgcn_wmma_f32_16x16x32_bf16(
                    false, Al.v, false, Bh.v, (short)0, c, false, false);
            c = __builtin_amdgcn_wmma_f32_16x16x32_bf16(
                    false, Ah.v, false, Bl.v, (short)0, c, false, false);
            c = __builtin_amdgcn_wmma_f32_16x16x32_bf16(
                    false, Ah.v, false, Bh.v, (short)0, c, false, false);

            if (cok) {
                const int f0 = base + mt * 16 + half16 * 8;  // C: VGPR r -> M=r (+8 hi half)
#pragma unroll
                for (int r = 0; r < 8; ++r)
                    out[(f0 + r) * NPTS + col] = c[r];
            }
        }
    }
}

extern "C" void kernel_launch(void* const* d_in, const int* in_sizes, int n_in,
                              void* d_out, int out_size, void* d_ws, size_t ws_size,
                              hipStream_t stream) {
    const float* x          = (const float*)d_in[0];
    const float* init_shape = (const float*)d_in[1];
    MlpW P;
    for (int l = 0; l < 8; ++l) {
        P.w[l] = (const float*)d_in[2 + 2 * l];
        P.b[l] = (const float*)d_in[3 + 2 * l];
    }
    const float* W9 = (const float*)d_in[18];
    float* out = (float*)d_out;
    const int nframes = in_sizes[0];

    unsigned short* w9hi = (unsigned short*)d_ws;                          // 64512 B
    unsigned short* w9lo = (unsigned short*)((char*)d_ws + W9_HALF_BYTES); // 64512 B

    const int prep_total = W9_ROWS_PAD * 32;
    prep_w9_kernel<<<(prep_total + TBF - 1) / TBF, TBF, 0, stream>>>(W9, w9hi, w9lo);

    const int nblocks = (nframes + TBF - 1) / TBF;
    shape_decoder_kernel<<<nblocks, TBF, 0, stream>>>(
        x, init_shape, P, (const uint4*)w9hi, (const uint4*)w9lo, out, nframes);
}
